// LinearAttention_82094004896542
// MI455X (gfx1250) — matrix operand order repaired and checked by reading, not compiled, not run on hardware
//
#include <hip/hip_runtime.h>

// ---------------------------------------------------------------------------
// Linear attention (chunked, parallel two-pass) for MI455X / gfx1250.
//   cvt/transpose -> qkv GEMM (bf16 WMMA, TDM double-buffered LDS panels)
//   -> pass A: per-chunk state G_c = v^T @ LN(k)      (4096 parallel blocks)
//   -> pass B: S_in[c] = prefix-sum of G               (memory only)
//   -> pass C: y = q @ S_in^T + tril(q k^T) @ v        (4096 parallel blocks)
//   -> output GEMM (bf16 WMMA, f32 out).
// All WMMA fragments are per-lane contiguous in LDS (ds_load_b128 path).
// ---------------------------------------------------------------------------

typedef __attribute__((ext_vector_type(16))) __bf16 v16bf;
typedef __attribute__((ext_vector_type(8)))  float  v8f;

#define LDSB 72   // bf16 LDS row stride (64 + 8 pad; 144B rows, 16B aligned)

__device__ __forceinline__ v8f wmma_bf16(v16bf a, v16bf b, v8f c) {
  return __builtin_amdgcn_wmma_f32_16x16x32_bf16(false, a, false, b, (short)0, c,
                                                 false, false);
}

// ---- WMMA fragment loaders (CDNA5 ISA 7.12.2 layouts, all contiguous) ------
// A (16x32): lane L -> row M=L&15; lanes<16: K={0..7,16..23}, lanes>=16: +8
__device__ __forceinline__ v16bf frag_a(const __bf16 (*m)[LDSB], int row0, int k0,
                                        int lane) {
  const __bf16* r = &m[row0 + (lane & 15)][0];
  int kb = k0 + ((lane & 16) >> 1);
  v16bf f;
#pragma unroll
  for (int i = 0; i < 8; ++i) f[i] = r[kb + i];
#pragma unroll
  for (int i = 0; i < 8; ++i) f[8 + i] = r[kb + 16 + i];
  return f;
}

// B (32x16): lane L -> col N=L&15; lanes<16: K={0..7,16..23}, lanes>=16: +8
// Source stored transposed (N-major): element B(k,n) = m[n][k] -> contiguous.
__device__ __forceinline__ v16bf frag_bt(const __bf16 (*m)[LDSB], int k0, int col0,
                                         int lane) {
  const __bf16* r = &m[col0 + (lane & 15)][0];
  int kb = k0 + ((lane & 16) >> 1);
  v16bf f;
#pragma unroll
  for (int i = 0; i < 8; ++i) f[i] = r[kb + i];
#pragma unroll
  for (int i = 0; i < 8; ++i) f[8 + i] = r[kb + 16 + i];
  return f;
}

// ---------------------------------------------------------------------------
// Tensor Data Mover staging (gfx1250). 2-D tile load, bf16, LDS pad so that
// the DMA writes rows with stride 72 bf16 (pad 4 DWORDs every 32 DWORDs).
// ---------------------------------------------------------------------------
#if __has_builtin(__builtin_amdgcn_tensor_load_to_lds) && \
    __has_builtin(__builtin_amdgcn_s_wait_tensorcnt)
#define USE_TDM 1
typedef __attribute__((ext_vector_type(4))) unsigned int v4u;
typedef __attribute__((ext_vector_type(8))) int         v8i_t;
typedef __attribute__((ext_vector_type(4))) int         v4i_t;

__device__ __forceinline__ void tdm_load_tile_2d(const __bf16* lds_dst,
                                                 const __bf16* gsrc,
                                                 unsigned tile_k,     // dim0 elems
                                                 unsigned tile_rows,  // dim1
                                                 unsigned row_stride) // elems
{
  unsigned long long ga = (unsigned long long)gsrc;
  unsigned lds_off = (unsigned)(unsigned long long)lds_dst;  // low 32b = LDS offset
  v4u g0;
  g0[0] = 1u;                                                // count=1, user D#
  g0[1] = lds_off;                                           // lds_addr
  g0[2] = (unsigned)ga;                                      // global_addr lo
  g0[3] = (unsigned)((ga >> 32) & 0x01ffffffu) | (2u << 30); // addr hi | type=2
  v8i_t g1;
  g1[0] = (int)((1u << 16)      // data_size = 2 bytes
              | (1u << 20)      // pad_enable
              | (4u << 22)      // pad_interval: 32 DWORDs (one 64-elem bf16 row)
              | (3u << 25));    // pad_amount: 4 DWORDs (8 bf16 -> stride 72)
  g1[1] = 0;                                       // atomic_addr | tensor_dim0 lo16
  g1[2] = (int)0x0400u;                            // tensor_dim0 hi16 | dim1 lo16
  g1[3] = (int)(0x0400u | (tile_k << 16));         // tensor_dim1 hi16 | tile_dim0
  g1[4] = (int)(tile_rows & 0xffffu);              // tile_dim1 | tile_dim2=0
  g1[5] = (int)row_stride;                         // tensor_dim0_stride lo32
  g1[6] = 0;
  g1[7] = 0;
  v4i_t z4 = {0, 0, 0, 0};
#if defined(__clang_major__) && __clang_major__ >= 23
  v8i_t z8 = {0, 0, 0, 0, 0, 0, 0, 0};
  __builtin_amdgcn_tensor_load_to_lds(g0, g1, z4, z4, z8, 0);
#else
  __builtin_amdgcn_tensor_load_to_lds(g0, g1, z4, z4, 0);
#endif
}
#else
#define USE_TDM 0
#endif

// ---------------------------------------------------------------------------
// f32 -> bf16 conversion (4-wide)
// ---------------------------------------------------------------------------
__global__ __launch_bounds__(256) void cvt_f32_bf16(const float4* __restrict__ s,
                                                    __bf16* __restrict__ d, int n4) {
  int i = blockIdx.x * blockDim.x + threadIdx.x;
  if (i < n4) {
    float4 v = s[i];
    int j = i << 2;
    d[j + 0] = (__bf16)v.x;
    d[j + 1] = (__bf16)v.y;
    d[j + 2] = (__bf16)v.z;
    d[j + 3] = (__bf16)v.w;
  }
}

// f32 RxC  ->  bf16 CxR (transposed), 32x32 LDS tiles
__global__ __launch_bounds__(256) void cvt_transpose(const float* __restrict__ s,
                                                     __bf16* __restrict__ d,
                                                     int R, int C) {
  __shared__ float t[32][33];
  int c0 = blockIdx.x * 32, r0 = blockIdx.y * 32;
  int lx = threadIdx.x & 31, ly = threadIdx.x >> 5;
#pragma unroll
  for (int i = 0; i < 32; i += 8)
    t[ly + i][lx] = s[(size_t)(r0 + ly + i) * C + c0 + lx];
  __syncthreads();
#pragma unroll
  for (int i = 0; i < 32; i += 8)
    d[(size_t)(c0 + ly + i) * R + r0 + lx] = (__bf16)t[lx][ly + i];
}

// ---------------------------------------------------------------------------
// Tiled bf16 WMMA GEMM: C(MxN) = A(MxK) @ B(KxN), B supplied TRANSPOSED (NxK).
// 128x64 block tile, 8 waves, 2x2 16x16 tiles per wave.
// TDM double-buffered K-panels: DMA for panel i+1 overlaps WMMA on panel i
// (in-order TENSORcnt retirement => s_wait_tensorcnt(1) pins current panel).
// bf16 epilogue repacks through LDS into global_store_b128.
// ---------------------------------------------------------------------------
template <int F32OUT>
__global__ __launch_bounds__(256) void gemm128(const __bf16* __restrict__ A,
                                               const __bf16* __restrict__ Bt,
                                               __bf16* __restrict__ Cb,
                                               float* __restrict__ Cf,
                                               int N, int K) {
  __shared__ __bf16 As[2][128][LDSB];
  __shared__ __bf16 Bs[2][64][LDSB];   // [n][k]
  const int tid = threadIdx.x, lane = tid & 31, w = tid >> 5;
  const int row0 = blockIdx.x * 128, col0 = blockIdx.y * 64;
  const int tr0 = (w >> 1) << 5;    // 0,32,64,96
  const int tc0 = (w & 1) << 5;     // 0,32
  v8f acc00 = {}, acc01 = {}, acc10 = {}, acc11 = {};

  int buf = 0;
#if USE_TDM
  if (w == 0)
    tdm_load_tile_2d(&As[0][0][0], A + (size_t)row0 * K, 64u, 128u, (unsigned)K);
  else if (w == 1)
    tdm_load_tile_2d(&Bs[0][0][0], Bt + (size_t)col0 * K, 64u, 64u, (unsigned)K);
#endif
  for (int k0 = 0; k0 < K; k0 += 64) {
#if USE_TDM
    if (k0 + 64 < K) {  // prefetch next panel into the other buffer
      if (w == 0)
        tdm_load_tile_2d(&As[buf ^ 1][0][0], A + (size_t)row0 * K + k0 + 64,
                         64u, 128u, (unsigned)K);
      else if (w == 1)
        tdm_load_tile_2d(&Bs[buf ^ 1][0][0], Bt + (size_t)col0 * K + k0 + 64,
                         64u, 64u, (unsigned)K);
      __builtin_amdgcn_s_wait_tensorcnt(1);   // current panel resident
    } else {
      __builtin_amdgcn_s_wait_tensorcnt(0);
    }
    __syncthreads();
#else
    {
      const int ar = tid >> 1, ac = (tid & 1) << 5;
      const __bf16* ag = A + (size_t)(row0 + ar) * K + k0 + ac;
      *(uint4*)&As[0][ar][ac]      = *(const uint4*)(ag);
      *(uint4*)&As[0][ar][ac + 8]  = *(const uint4*)(ag + 8);
      *(uint4*)&As[0][ar][ac + 16] = *(const uint4*)(ag + 16);
      *(uint4*)&As[0][ar][ac + 24] = *(const uint4*)(ag + 24);
      const int br = tid >> 2, bc = (tid & 3) << 4;
      const __bf16* bg = Bt + (size_t)(col0 + br) * K + k0 + bc;
      *(uint4*)&Bs[0][br][bc]     = *(const uint4*)(bg);
      *(uint4*)&Bs[0][br][bc + 8] = *(const uint4*)(bg + 8);
      if (k0 + 64 < K) {
        __builtin_prefetch(ag + 64, 0, 0);
        __builtin_prefetch(bg + 64, 0, 0);
      }
    }
    __syncthreads();
#endif
#pragma unroll
    for (int ks = 0; ks < 64; ks += 32) {
      v16bf a0 = frag_a(As[buf], tr0, ks, lane);
      v16bf a1 = frag_a(As[buf], tr0 + 16, ks, lane);
      v16bf b0 = frag_bt(Bs[buf], ks, tc0, lane);
      v16bf b1 = frag_bt(Bs[buf], ks, tc0 + 16, lane);
      acc00 = wmma_bf16(a0, b0, acc00);
      acc01 = wmma_bf16(a0, b1, acc01);
      acc10 = wmma_bf16(a1, b0, acc10);
      acc11 = wmma_bf16(a1, b1, acc11);
    }
    __syncthreads();
#if USE_TDM
    buf ^= 1;
#endif
  }

  const int hi8 = (lane & 16) >> 1;
  const int cn = lane & 15;
  if (F32OUT) {
#pragma unroll
    for (int i = 0; i < 8; ++i) {
      int r0_ = row0 + tr0 + hi8 + i;
      int r1_ = r0_ + 16;
      int c0_ = col0 + tc0 + cn;
      int c1_ = c0_ + 16;
      Cf[(size_t)r0_ * N + c0_] = acc00[i];
      Cf[(size_t)r0_ * N + c1_] = acc01[i];
      Cf[(size_t)r1_ * N + c0_] = acc10[i];
      Cf[(size_t)r1_ * N + c1_] = acc11[i];
    }
  } else {
    // repack 128x64 bf16 output block through LDS -> global_store_b128
    __bf16 (*Ep)[LDSB] = As[0];
#pragma unroll
    for (int i = 0; i < 8; ++i) {
      Ep[tr0 + hi8 + i][tc0 + cn]           = (__bf16)acc00[i];
      Ep[tr0 + hi8 + i][tc0 + 16 + cn]      = (__bf16)acc01[i];
      Ep[tr0 + 16 + hi8 + i][tc0 + cn]      = (__bf16)acc10[i];
      Ep[tr0 + 16 + hi8 + i][tc0 + 16 + cn] = (__bf16)acc11[i];
    }
    __syncthreads();
    const int er = tid >> 1, ec = (tid & 1) << 5;
    __bf16* cg = Cb + (size_t)(row0 + er) * N + col0 + ec;
    *(uint4*)(cg)      = *(const uint4*)&Ep[er][ec];
    *(uint4*)(cg + 8)  = *(const uint4*)&Ep[er][ec + 8];
    *(uint4*)(cg + 16) = *(const uint4*)&Ep[er][ec + 16];
    *(uint4*)(cg + 24) = *(const uint4*)&Ep[er][ec + 24];
  }
}

// ---------------------------------------------------------------------------
// Pass A: per-chunk state contribution  G_c = v_c^T @ LN(k_c)   (d x e)
// One block per (b,h,chunk) = 4096 blocks.  Raw-k tile staged by TDM.
// ---------------------------------------------------------------------------
__global__ __launch_bounds__(256) void attn_chunk_state(const __bf16* __restrict__ qkv,
                                                        __bf16* __restrict__ G) {
  __shared__ __bf16 kt[64][LDSB];    // raw k, t-major (for LN)
  __shared__ __bf16 knt[64][LDSB];   // normalized k, e-major  [e][t]
  __shared__ __bf16 vt[64][LDSB];    // v, d-major             [d][t]
  __shared__ float lmu[64], lrs[64];
  const int tid = threadIdx.x, lane = tid & 31, w = tid >> 5;
  const int bhc = blockIdx.x;
  const int bh = bhc >> 6, ch = bhc & 63;
  const int b = bh >> 4, h = bh & 15;
  const int lr = tid >> 2, lc = (tid & 3) << 4;
  const size_t tok0 = (size_t)(b * 4096 + ch * 64);
  const __bf16* base = qkv + (tok0 + lr) * 3072 + h * 64 + lc;

#if USE_TDM
  if (w == 0) {
    tdm_load_tile_2d(&kt[0][0], qkv + tok0 * 3072 + h * 64 + 1024, 64u, 64u, 3072u);
    __builtin_amdgcn_s_wait_tensorcnt(0);
  }
#else
  *(uint4*)&kt[lr][lc]     = *(const uint4*)(base + 1024);
  *(uint4*)&kt[lr][lc + 8] = *(const uint4*)(base + 1032);
#endif
  {  // v: transposed scatter (TDM cannot transpose)
    uint4 u0 = *(const uint4*)(base + 2048);
    uint4 u1 = *(const uint4*)(base + 2056);
    const __bf16* p0 = (const __bf16*)&u0;
    const __bf16* p1 = (const __bf16*)&u1;
#pragma unroll
    for (int i = 0; i < 8; ++i) vt[lc + i][lr] = p0[i];
#pragma unroll
    for (int i = 0; i < 8; ++i) vt[lc + 8 + i][lr] = p1[i];
  }
  __syncthreads();
  if (tid < 64) {  // row LN statistics
    float mu = 0.f;
#pragma unroll
    for (int i = 0; i < 64; ++i) mu += (float)kt[tid][i];
    mu *= (1.f / 64.f);
    float var = 0.f;
#pragma unroll
    for (int i = 0; i < 64; ++i) { float dd = (float)kt[tid][i] - mu; var += dd * dd; }
    lmu[tid] = mu;
    lrs[tid] = rsqrtf(var * (1.f / 64.f) + 1e-5f);
  }
  __syncthreads();
  {  // normalize + transpose scatter -> knt[e][t]
    float mu = lmu[lr], rs = lrs[lr];
#pragma unroll
    for (int i = 0; i < 16; ++i)
      knt[lc + i][lr] = (__bf16)(((float)kt[lr][lc + i] - mu) * rs);
  }
  __syncthreads();

  const int mt = w >> 1, nt0 = (w & 1) << 1;
  const int hi8 = (lane & 16) >> 1, cn = lane & 15;
  __bf16* Gc = G + (size_t)bhc * 4096;
#pragma unroll
  for (int t2 = 0; t2 < 2; ++t2) {
    int nt = nt0 + t2;
    v8f c = {};
#pragma unroll
    for (int ks0 = 0; ks0 < 64; ks0 += 32) {
      v16bf a  = frag_a(vt, mt * 16, ks0, lane);     // A(d,t) = vt[d][t]
      v16bf bb = frag_bt(knt, ks0, nt * 16, lane);   // B(t,e) = knt[e][t]
      c = wmma_bf16(a, bb, c);
    }
    int e = nt * 16 + cn;
#pragma unroll
    for (int i = 0; i < 8; ++i)
      Gc[(mt * 16 + hi8 + i) * 64 + e] = (__bf16)c[i];
  }
}

// ---------------------------------------------------------------------------
// Pass B: exclusive prefix over chunks:  S_in[c] = sum_{j<c} G[j]  per (b,h).
// Memory-only; 64 blocks, each thread owns 16 consecutive state elements.
// ---------------------------------------------------------------------------
__global__ __launch_bounds__(256) void state_prefix(const __bf16* __restrict__ G,
                                                    __bf16* __restrict__ Sin) {
  const int bh = blockIdx.x;
  const int base = threadIdx.x * 16;
  float acc[16];
#pragma unroll
  for (int i = 0; i < 16; ++i) acc[i] = 0.f;
  for (int c = 0; c < 64; ++c) {
    size_t off = ((size_t)bh * 64 + c) * 4096 + base;
    __bf16 tmp[16];
#pragma unroll
    for (int i = 0; i < 16; ++i) tmp[i] = (__bf16)acc[i];
    *(uint4*)(Sin + off)     = *(const uint4*)&tmp[0];
    *(uint4*)(Sin + off + 8) = *(const uint4*)&tmp[8];
    uint4 g0 = *(const uint4*)(G + off);
    uint4 g1 = *(const uint4*)(G + off + 8);
    const __bf16* p0 = (const __bf16*)&g0;
    const __bf16* p1 = (const __bf16*)&g1;
#pragma unroll
    for (int i = 0; i < 8; ++i) acc[i] += (float)p0[i];
#pragma unroll
    for (int i = 0; i < 8; ++i) acc[8 + i] += (float)p1[i];
  }
}

// ---------------------------------------------------------------------------
// Pass C: y_c = q_c @ S_in[c]^T + tril(q_c @ k_n^T) @ v_c.  4096 blocks.
// q / raw-k / S_in tiles staged by TDM; v staged via transposed scatter.
// ---------------------------------------------------------------------------
__global__ __launch_bounds__(256) void attn_output(const __bf16* __restrict__ qkv,
                                                   const __bf16* __restrict__ Sin,
                                                   __bf16* __restrict__ y) {
  __shared__ __bf16 qs[64][LDSB];   // q, t-major
  __shared__ __bf16 kn[64][LDSB];   // k (raw then normalized), t-major
  __shared__ __bf16 vt[64][LDSB];   // v, d-major
  __shared__ __bf16 Ss[64][LDSB];   // S_in, d-major [d][e]
  __shared__ __bf16 ss[64][LDSB];   // scores, t-major
  __shared__ float lmu[64], lrs[64];
  const int tid = threadIdx.x, lane = tid & 31, w = tid >> 5;
  const int bhc = blockIdx.x;
  const int bh = bhc >> 6, ch = bhc & 63;
  const int b = bh >> 4, h = bh & 15;
  const int lr = tid >> 2, lc = (tid & 3) << 4;
  const size_t tok0 = (size_t)(b * 4096 + ch * 64);
  const __bf16* base = qkv + (tok0 + lr) * 3072 + h * 64 + lc;

#if USE_TDM
  if (w == 0) {
    tdm_load_tile_2d(&qs[0][0], qkv + tok0 * 3072 + h * 64,        64u, 64u, 3072u);
    tdm_load_tile_2d(&kn[0][0], qkv + tok0 * 3072 + h * 64 + 1024, 64u, 64u, 3072u);
    tdm_load_tile_2d(&Ss[0][0], Sin + (size_t)bhc * 4096,          64u, 64u, 64u);
    __builtin_amdgcn_s_wait_tensorcnt(0);
  }
#else
  {
    *(uint4*)&qs[lr][lc]     = *(const uint4*)(base);
    *(uint4*)&qs[lr][lc + 8] = *(const uint4*)(base + 8);
    *(uint4*)&kn[lr][lc]     = *(const uint4*)(base + 1024);
    *(uint4*)&kn[lr][lc + 8] = *(const uint4*)(base + 1032);
    const __bf16* sb = Sin + (size_t)bhc * 4096 + lr * 64 + lc;
    *(uint4*)&Ss[lr][lc]     = *(const uint4*)(sb);
    *(uint4*)&Ss[lr][lc + 8] = *(const uint4*)(sb + 8);
  }
#endif
  {  // v: transposed scatter
    uint4 u0 = *(const uint4*)(base + 2048);
    uint4 u1 = *(const uint4*)(base + 2056);
    const __bf16* p0 = (const __bf16*)&u0;
    const __bf16* p1 = (const __bf16*)&u1;
#pragma unroll
    for (int i = 0; i < 8; ++i) vt[lc + i][lr] = p0[i];
#pragma unroll
    for (int i = 0; i < 8; ++i) vt[lc + 8 + i][lr] = p1[i];
  }
  __syncthreads();
  if (tid < 64) {
    float mu = 0.f;
#pragma unroll
    for (int i = 0; i < 64; ++i) mu += (float)kn[tid][i];
    mu *= (1.f / 64.f);
    float var = 0.f;
#pragma unroll
    for (int i = 0; i < 64; ++i) { float dd = (float)kn[tid][i] - mu; var += dd * dd; }
    lmu[tid] = mu;
    lrs[tid] = rsqrtf(var * (1.f / 64.f) + 1e-5f);
  }
  __syncthreads();
  {  // normalize k in place (contiguous)
    float mu = lmu[lr], rs = lrs[lr];
#pragma unroll
    for (int i = 0; i < 16; ++i)
      kn[lr][lc + i] = (__bf16)(((float)kn[lr][lc + i] - mu) * rs);
  }
  __syncthreads();

  const int mt = w >> 1, nt0 = (w & 1) << 1;
  const int hi8 = (lane & 16) >> 1, cn = lane & 15;

  // scores = tril(q @ k^T)   (t x s)
#pragma unroll
  for (int t2 = 0; t2 < 2; ++t2) {
    int nt = nt0 + t2;
    v8f c = {};
#pragma unroll
    for (int ks0 = 0; ks0 < 64; ks0 += 32) {
      v16bf a  = frag_a(qs, mt * 16, ks0, lane);
      v16bf bb = frag_bt(kn, ks0, nt * 16, lane);   // B(d,s) = kn[s][d]
      c = wmma_bf16(a, bb, c);
    }
    int s = nt * 16 + cn;
#pragma unroll
    for (int i = 0; i < 8; ++i) {
      int t = mt * 16 + hi8 + i;
      ss[t][s] = (__bf16)((s <= t) ? c[i] : 0.f);
    }
  }
  __syncthreads();

  // y = q @ S_in^T + scores @ v   (t x d)
#pragma unroll
  for (int t2 = 0; t2 < 2; ++t2) {
    int nt = nt0 + t2;
    v8f c = {};
#pragma unroll
    for (int ks0 = 0; ks0 < 64; ks0 += 32) {
      v16bf a  = frag_a(qs, mt * 16, ks0, lane);
      v16bf bb = frag_bt(Ss, ks0, nt * 16, lane);   // B(e,d) = Ss[d][e]
      c = wmma_bf16(a, bb, c);
      v16bf a2 = frag_a(ss, mt * 16, ks0, lane);
      v16bf b2 = frag_bt(vt, ks0, nt * 16, lane);   // B(s,d) = vt[d][s]
      c = wmma_bf16(a2, b2, c);
    }
    int d = nt * 16 + cn;
#pragma unroll
    for (int i = 0; i < 8; ++i) {
      int t = mt * 16 + hi8 + i;
      y[(tok0 + t) * 1024 + h * 64 + d] = (__bf16)c[i];
    }
  }
}

// ---------------------------------------------------------------------------
// Host-side orchestration
// ---------------------------------------------------------------------------
extern "C" void kernel_launch(void* const* d_in, const int* in_sizes, int n_in,
                              void* d_out, int out_size, void* d_ws, size_t ws_size,
                              hipStream_t stream) {
  (void)in_sizes; (void)n_in; (void)out_size; (void)ws_size;
  const float* x    = (const float*)d_in[0];   // (4,4096,1024)
  const float* Wqkv = (const float*)d_in[1];   // (1024,3072)
  const float* Wo   = (const float*)d_in[2];   // (1024,1024)
  float* out = (float*)d_out;                  // (4,4096,1024)

  // Workspace layout (bytes, 16B aligned)
  char* w = (char*)d_ws;
  __bf16* xb     = (__bf16*)(w);                 // 32 MB, reused as y after GEMM1
  __bf16* wqkvT  = (__bf16*)(w + 33554432);      //  6 MB  (3072 x 1024, transposed)
  __bf16* woT    = (__bf16*)(w + 39845888);      //  2 MB  (1024 x 1024, transposed)
  __bf16* qkvb   = (__bf16*)(w + 41943040);      // 96 MB  (16384 x 3072)
  __bf16* Gbuf   = (__bf16*)(w + 142606336);     // 32 MB  (4096 x 64 x 64)
  __bf16* Sin    = (__bf16*)(w + 176160768);     // 32 MB  (4096 x 64 x 64)
  __bf16* yb     = xb;

  // 1) conversions (weights converted + transposed so GEMM B-panels are N-major)
  cvt_f32_bf16<<<16384, 256, 0, stream>>>((const float4*)x, xb, 4194304);
  cvt_transpose<<<dim3(96, 32), 256, 0, stream>>>(Wqkv, wqkvT, 1024, 3072);
  cvt_transpose<<<dim3(32, 32), 256, 0, stream>>>(Wo, woT, 1024, 1024);

  // 2) qkv = x @ W_qkv   (M=16384, N=3072, K=1024)
  gemm128<0><<<dim3(128, 48), 256, 0, stream>>>(xb, wqkvT, qkvb, nullptr, 3072, 1024);

  // 3) attention: parallel chunked form
  attn_chunk_state<<<4096, 256, 0, stream>>>(qkvb, Gbuf);
  state_prefix<<<64, 256, 0, stream>>>(Gbuf, Sin);
  attn_output<<<4096, 256, 0, stream>>>(qkvb, Sin, yb);

  // 4) out = y @ W_o     (M=16384, N=1024, K=1024)
  gemm128<1><<<dim3(128, 16), 256, 0, stream>>>(yb, woT, nullptr, out, 1024, 1024);
}
